// CausalSelfAttention_58892591563183
// MI455X (gfx1250) — compile-verified, matrix operand order checked by reading, not run on hardware
//
#include <hip/hip_runtime.h>

// ---------------------------------------------------------------------------
// Causal self-attention for MI455X (gfx1250, wave32, WMMA).
// B=4, T=2048, C=1024, H=16, D=64, F=3*C=3072, M=B*T=8192.
// bf16 storage / f32 accumulation via v_wmma_f32_16x16x32_bf16.
// GEMM A tiles staged via GLOBAL_LOAD_ASYNC_TO_LDS_B128 (ASYNCcnt path);
// GEMM B tiles staged via the Tensor Data Mover (TENSOR_LOAD_TO_LDS,
// TENSORcnt path) when the builtin exists, else async copies.
// ---------------------------------------------------------------------------

#define BATCH 4
#define SEQ   2048
#define DMODEL 1024
#define NHEADS 16
#define HDIM  64
#define FQKV  3072
#define MTOK  (BATCH * SEQ)          // 8192

typedef __attribute__((ext_vector_type(16))) __bf16 v16bf;
typedef __attribute__((ext_vector_type(8)))  __bf16 v8bf;
typedef __attribute__((ext_vector_type(8)))  float  v8f;
typedef __attribute__((ext_vector_type(4)))  int    v4i;
typedef __attribute__((ext_vector_type(4)))  unsigned int v4u;
typedef __attribute__((ext_vector_type(8)))  int    v8i;

#if defined(__has_builtin)
#if __has_builtin(__builtin_amdgcn_global_load_async_to_lds_b128)
#define HAVE_ASYNC_LDS 1
#endif
#if __has_builtin(__builtin_amdgcn_s_wait_asynccnt)
#define HAVE_WAIT_ASYNC 1
#endif
#if __has_builtin(__builtin_amdgcn_tensor_load_to_lds) && \
    __has_builtin(__builtin_amdgcn_s_wait_tensorcnt)
#define HAVE_TDM 1
#endif
#endif
#ifndef HAVE_ASYNC_LDS
#define HAVE_ASYNC_LDS 0
#endif
#ifndef HAVE_WAIT_ASYNC
#define HAVE_WAIT_ASYNC 0
#endif
#ifndef HAVE_TDM
#define HAVE_TDM 0
#endif

__device__ __forceinline__ v8f wmma_bf16(v16bf a, v16bf b, v8f c) {
  // (neg_a, A, neg_b, B, c_mod, C, reuse_a, reuse_b)
  return __builtin_amdgcn_wmma_f32_16x16x32_bf16(false, a, false, b, (short)0, c,
                                                 false, false);
}

// 16-byte global -> LDS copy. Async (ASYNCcnt) path when available.
// Probe-derived signature: (v4i* src, v4i* dst, int offset, int cpol).
__device__ __forceinline__ void copy16_g2l(const __bf16* g, __bf16* l) {
#if HAVE_ASYNC_LDS
  __builtin_amdgcn_global_load_async_to_lds_b128((v4i*)g, (v4i*)l, 0, 0);
#else
  *(v8bf*)l = *(const v8bf*)g;   // global_load_b128 + ds_store_b128
#endif
}

__device__ __forceinline__ void async_wait0() {
#if HAVE_ASYNC_LDS
#if HAVE_WAIT_ASYNC
  __builtin_amdgcn_s_wait_asynccnt(0);
#else
  asm volatile("s_wait_asynccnt 0" ::: "memory");
#endif
#endif
}

#if HAVE_TDM
// TDM DMA: load a [rows x 32] bf16 tile (row stride `ld` elements) from
// global memory into LDS. D# bitfields per CDNA5 ISA ch.8 (08_async_tensor):
//   group0: [1:0]=count=1, [63:32]=lds_addr, [120:64]=global_addr,
//           [127:126]=type=2
//   group1: [17:16]=data_size(1=2B), [79:48]=tensor_dim0, [111:80]=tensor_dim1,
//           [127:112]=tile_dim0, [143:128]=tile_dim1, [207:160]=dim0_stride
//   groups 2/3: zero (<=2-D tensor)
// This toolchain's builtin is the 6-arg form:
//   (uint32x4 g0, int32x8 g1, int32x4, int32x4, int32x8, i32 cpol)
__device__ __forceinline__ void tdm_load_tile_b(const __bf16* g, __bf16* l,
                                                int ld, int rows, int K) {
  const unsigned long long ga = (unsigned long long)(const void*)g;
  const unsigned lds = (unsigned)(size_t)l;          // LDS byte offset (addr[31:0])
  const unsigned td0 = (unsigned)K;                  // tensor extent, dim0
  const unsigned td1 = 1u << 20;                     // generous dim1 extent
  v4u g0 = {1u, lds, (unsigned)ga,
            (unsigned)((ga >> 32) & 0x01FFFFFFull) | 0x80000000u};
  v8i g1 = {(int)(1u << 16),                                   // data_size = 2B
            (int)((td0 & 0xFFFFu) << 16),                      // dim0 [15:0]
            (int)(((td0 >> 16) & 0xFFFFu) | ((td1 & 0xFFFFu) << 16)),
            (int)(((td1 >> 16) & 0xFFFFu) | (32u << 16)),      // tile_dim0 = 32
            (int)(unsigned)rows,                               // tile_dim1
            (int)(unsigned)ld,                                 // dim0 stride lo32
            0, 0};
  v4i gz = {0, 0, 0, 0};
  v8i gz8 = {0, 0, 0, 0, 0, 0, 0, 0};
  __builtin_amdgcn_tensor_load_to_lds(g0, g1, gz, gz, gz8, 0);
}
#endif

// Load one 16x32 bf16 WMMA operand fragment from a K-major matrix in GLOBAL
// memory (row stride `ld` elements, K contiguous). Works for both A operands
// (row = M index) and B operands (row = N index), matching the ISA VGPR
// layout: lane L holds row (L&15); half h=L>>4 holds K = h*8..h*8+7 in
// elements 0..7 and K = 16+h*8..16+h*8+7 in elements 8..15.
__device__ __forceinline__ v16bf load_frag_g(const __bf16* __restrict__ base,
                                             int row0, int ld, int k0, int lane) {
  const int r = row0 + (lane & 15);
  const int h = (lane >> 4) & 1;
  const __bf16* p = base + (size_t)r * (size_t)ld + (size_t)(k0 + h * 8);
  v8bf lo = *(const v8bf*)p;          // K = k0+h*8 .. +7        (global_load_b128)
  v8bf hi = *(const v8bf*)(p + 16);   // K = k0+16+h*8 .. +7
  v16bf f;
#pragma unroll
  for (int i = 0; i < 8; ++i) { f[i] = lo[i]; f[i + 8] = hi[i]; }
  return f;
}

// Same fragment pattern, but from an LDS tile with row stride 32 elements.
__device__ __forceinline__ v16bf load_frag_lds(const __bf16* base, int row0,
                                               int lane) {
  const int r = row0 + (lane & 15);
  const int h = (lane >> 4) & 1;
  const __bf16* p = base + r * 32 + h * 8;
  v8bf lo = *(const v8bf*)p;          // ds_load_b128
  v8bf hi = *(const v8bf*)(p + 16);
  v16bf f;
#pragma unroll
  for (int i = 0; i < 8; ++i) { f[i] = lo[i]; f[i + 8] = hi[i]; }
  return f;
}

// ---------------------------------------------------------------------------
// Kernel 1: f32 -> bf16 conversion
// ---------------------------------------------------------------------------
__global__ void cvt_f32_bf16(const float* __restrict__ in, __bf16* __restrict__ out,
                             int n) {
  int i = blockIdx.x * blockDim.x + threadIdx.x;
  if (i < n) out[i] = (__bf16)in[i];
}

// ---------------------------------------------------------------------------
// Kernel 2: NT GEMM  C[M,N] = A[M,K] * B[N,K]^T   (both operands K-major bf16)
// OUT_MODE 0: bf16 output, 0.125 scale folded into columns n<1024 (Q cols)
// OUT_MODE 1: f32 output
// Block: 256 threads = 8 waves, 4(M) x 2(N) wave tiles of 32x32; block tile
// 128x64, K-step 32. A tile (8KB) via per-lane async B128 copies; B tile
// (4KB) via one TDM descriptor issued by wave 0 (TENSORcnt).
// ---------------------------------------------------------------------------
template <int OUT_MODE>
__global__ __launch_bounds__(256) void gemm_bf16nt(
    const __bf16* __restrict__ A, const __bf16* __restrict__ Bw,
    void* __restrict__ Cout, int K, int lda, int ldb, int ldc) {
  __shared__ __align__(16) __bf16 LdsA[128][32];   // 8 KB
  __shared__ __align__(16) __bf16 LdsB[64][32];    // 4 KB

  const int tid = threadIdx.x;
  const int lane = tid & 31;
  const int wave = tid >> 5;
  const int m0 = (wave >> 1) * 32;                 // wave tile (block-local)
  const int n0 = (wave & 1) * 32;
  const size_t blockM = (size_t)blockIdx.y * 128;
  const size_t blockN = (size_t)blockIdx.x * 64;

  // Per-thread staging assignment: chunk = 16 bytes = 8 bf16.
  const int arow0 = tid >> 2;                      // A chunks 0..255
  const int arow1 = (tid + 256) >> 2;              // A chunks 256..511
  const int akc0 = (tid & 3) * 8;
  const int brow = tid >> 2;                       // B chunks 0..255 (fallback)
  const int bkc = (tid & 3) * 8;
  (void)brow; (void)bkc;

  v8f acc00 = {}, acc01 = {}, acc10 = {}, acc11 = {};
  for (int k0 = 0; k0 < K; k0 += 32) {
    __syncthreads();   // previous iteration finished reading the LDS tiles
    copy16_g2l(A + (blockM + arow0) * lda + k0 + akc0, &LdsA[arow0][akc0]);
    copy16_g2l(A + (blockM + arow1) * lda + k0 + akc0, &LdsA[arow1][akc0]);
#if HAVE_TDM
    if (wave == 0) {
      tdm_load_tile_b(Bw + blockN * ldb + k0, &LdsB[0][0], ldb, 64, K);
    }
#else
    copy16_g2l(Bw + (blockN + brow) * ldb + k0 + bkc, &LdsB[brow][bkc]);
#endif
    if (k0 + 32 < K) {  // prefetch next K-tile into cache (global_prefetch_b8)
      __builtin_prefetch(A + (blockM + arow0) * lda + k0 + 32 + akc0, 0, 1);
    }
    async_wait0();
#if HAVE_TDM
    if (wave == 0) __builtin_amdgcn_s_wait_tensorcnt(0);
#endif
    __syncthreads();

    v16bf a0 = load_frag_lds(&LdsA[0][0], m0, lane);
    v16bf a1 = load_frag_lds(&LdsA[0][0], m0 + 16, lane);
    v16bf b0 = load_frag_lds(&LdsB[0][0], n0, lane);
    v16bf b1 = load_frag_lds(&LdsB[0][0], n0 + 16, lane);
    acc00 = wmma_bf16(a0, b0, acc00);
    acc01 = wmma_bf16(a0, b1, acc01);
    acc10 = wmma_bf16(a1, b0, acc10);
    acc11 = wmma_bf16(a1, b1, acc11);
  }

  const int col = lane & 15;
  const int h = lane >> 4;
  auto store_tile = [&](v8f acc, int mt, int nt) {
    const int n = (int)blockN + n0 + nt * 16 + col;
#pragma unroll
    for (int r = 0; r < 8; ++r) {
      const size_t mrow = blockM + m0 + mt * 16 + r + 8 * h;  // C/D: row = r+8h
      if (OUT_MODE == 0) {
        const float sc = (n < DMODEL) ? 0.125f : 1.0f;  // 1/sqrt(HDIM) on Q
        ((__bf16*)Cout)[mrow * ldc + n] = (__bf16)(acc[r] * sc);
      } else {
        ((float*)Cout)[mrow * ldc + n] = acc[r];
      }
    }
  };
  store_tile(acc00, 0, 0);
  store_tile(acc01, 0, 1);
  store_tile(acc10, 1, 0);
  store_tile(acc11, 1, 1);
}

// ---------------------------------------------------------------------------
// Kernel 3: V transpose  QKV[b*T+t][2048 + h*64 + d] -> Vt[(bh*64+d)][t]
// so the PV B-operand has contiguous K (= key index t).
// ---------------------------------------------------------------------------
__global__ void transpose_v(const __bf16* __restrict__ QKV, __bf16* __restrict__ Vt) {
  const int t = blockIdx.x * 256 + threadIdx.x;
  const int d = blockIdx.y;
  const int bh = blockIdx.z;
  const int b = bh >> 4, h = bh & 15;
  const __bf16 v =
      QKV[(size_t)(b * SEQ + t) * FQKV + 2 * DMODEL + h * HDIM + d];
  Vt[((size_t)bh * HDIM + d) * SEQ + t] = v;
}

// ---------------------------------------------------------------------------
// Kernel 4: causal flash attention.
// Grid: (T/64, B*H). Block: 128 threads = 4 waves; wave w owns q rows
// [q0, q0+16). Per iteration: 32 keys -> S (2x 16x16 f32 tiles, 4 WMMA),
// online softmax (f32 VALU + shfl reductions), P->bf16 via per-wave LDS
// tile (C-layout -> A-layout transpose), O += P*V (4 WMMA).
// ---------------------------------------------------------------------------
__global__ __launch_bounds__(128) void attn_fa(
    const __bf16* __restrict__ QKV, const __bf16* __restrict__ Vt,
    __bf16* __restrict__ Ob) {
  const int lane = threadIdx.x & 31;
  const int wave = threadIdx.x >> 5;
  const int bh = blockIdx.y;
  const int b = bh >> 4, h = bh & 15;
  const int q0 = blockIdx.x * 64 + wave * 16;

  const __bf16* Qb = QKV + (size_t)b * SEQ * FQKV + h * HDIM;          // rows=q, K=d
  const __bf16* Kb = Qb + DMODEL;                                       // rows=key, K=d
  const __bf16* Vtb = Vt + (size_t)bh * HDIM * SEQ;                     // rows=d, K=key

  const v16bf qf0 = load_frag_g(Qb, q0, FQKV, 0, lane);
  const v16bf qf1 = load_frag_g(Qb, q0, FQKV, 32, lane);

  v8f o0 = {}, o1 = {}, o2 = {}, o3 = {};
  float m[8], l[8];
#pragma unroll
  for (int r = 0; r < 8; ++r) { m[r] = -1e30f; l[r] = 0.0f; }

  __shared__ __align__(16) __bf16 Pb[4][16][32];   // per-wave P tile (1 KB each)

  const int col = lane & 15;
  const int hh = lane >> 4;
  const int kend = q0 + 16;                         // causal key bound (per wave)

  for (int j0 = 0; j0 < kend; j0 += 32) {
    // ---- S = Q * K^T (16 q-rows x 32 keys) ----
    v16bf kf0 = load_frag_g(Kb, j0, FQKV, 0, lane);
    v16bf kf1 = load_frag_g(Kb, j0, FQKV, 32, lane);
    v16bf kf2 = load_frag_g(Kb, j0 + 16, FQKV, 0, lane);
    v16bf kf3 = load_frag_g(Kb, j0 + 16, FQKV, 32, lane);
    v8f s0 = {}, s1 = {};
    s0 = wmma_bf16(qf0, kf0, s0);
    s0 = wmma_bf16(qf1, kf1, s0);
    s1 = wmma_bf16(qf0, kf2, s1);
    s1 = wmma_bf16(qf1, kf3, s1);

    // ---- causal mask + online softmax (rows spread: row = r + 8*hh) ----
#pragma unroll
    for (int r = 0; r < 8; ++r) {
      const int qrow = q0 + r + 8 * hh;
      float v0 = (j0 + col <= qrow) ? s0[r] : -1e30f;
      float v1 = (j0 + 16 + col <= qrow) ? s1[r] : -1e30f;
      float mx = fmaxf(v0, v1);
#pragma unroll
      for (int off = 1; off < 16; off <<= 1)
        mx = fmaxf(mx, __shfl_xor(mx, off, 32));
      const float nm = fmaxf(m[r], mx);
      const float alpha = __expf(m[r] - nm);
      const float p0 = __expf(v0 - nm);
      const float p1 = __expf(v1 - nm);
      float rs = p0 + p1;
#pragma unroll
      for (int off = 1; off < 16; off <<= 1) rs += __shfl_xor(rs, off, 32);
      l[r] = l[r] * alpha + rs;
      m[r] = nm;
      o0[r] *= alpha; o1[r] *= alpha; o2[r] *= alpha; o3[r] *= alpha;
      Pb[wave][r + 8 * hh][col] = (__bf16)p0;
      Pb[wave][r + 8 * hh][16 + col] = (__bf16)p1;
    }

    // Same-wave LDS RAW: make the P stores visible before the A-layout reload.
    asm volatile("s_wait_dscnt 0" ::: "memory");

    // ---- reload P in A-operand layout: lane row = lane&15 ----
    v16bf pf;
    {
      const __bf16* pp = &Pb[wave][lane & 15][hh * 8];
      v8bf lo = *(const v8bf*)pp;         // K = hh*8..+7        (ds_load_b128)
      v8bf hi = *(const v8bf*)(pp + 16);  // K = 16+hh*8..+7
#pragma unroll
      for (int i = 0; i < 8; ++i) { pf[i] = lo[i]; pf[i + 8] = hi[i]; }
    }

    // ---- O += P * V ----
    v16bf vf0 = load_frag_g(Vtb, 0, SEQ, j0, lane);
    v16bf vf1 = load_frag_g(Vtb, 16, SEQ, j0, lane);
    v16bf vf2 = load_frag_g(Vtb, 32, SEQ, j0, lane);
    v16bf vf3 = load_frag_g(Vtb, 48, SEQ, j0, lane);
    o0 = wmma_bf16(pf, vf0, o0);
    o1 = wmma_bf16(pf, vf1, o1);
    o2 = wmma_bf16(pf, vf2, o2);
    o3 = wmma_bf16(pf, vf3, o3);
  }

  // ---- epilogue: O /= l, store bf16 [B*T, C] at cols h*64 + d ----
#pragma unroll
  for (int r = 0; r < 8; ++r) {
    const float inv = 1.0f / l[r];
    const size_t row =
        (size_t)(b * SEQ + q0 + r + 8 * hh) * DMODEL + h * HDIM;
    Ob[row + 0 * 16 + col] = (__bf16)(o0[r] * inv);
    Ob[row + 1 * 16 + col] = (__bf16)(o1[r] * inv);
    Ob[row + 2 * 16 + col] = (__bf16)(o2[r] * inv);
    Ob[row + 3 * 16 + col] = (__bf16)(o3[r] * inv);
  }
}

// ---------------------------------------------------------------------------
// Host launcher
// ---------------------------------------------------------------------------
extern "C" void kernel_launch(void* const* d_in, const int* in_sizes, int n_in,
                              void* d_out, int out_size, void* d_ws, size_t ws_size,
                              hipStream_t stream) {
  (void)in_sizes; (void)n_in; (void)out_size; (void)ws_size;
  const float* x = (const float*)d_in[0];      // [4,2048,1024]
  const float* wqkv = (const float*)d_in[1];   // [3072,1024]
  const float* wout = (const float*)d_in[2];   // [1024,1024]
  float* out = (float*)d_out;                  // [4,2048,1024]

  // Workspace carving (bytes):
  char* ws = (char*)d_ws;
  __bf16* Xb  = (__bf16*)(ws + 0);            // 8192*1024*2  = 16 MiB
  __bf16* Wq  = (__bf16*)(ws + 16777216);     // 3072*1024*2  =  6 MiB
  __bf16* Wo  = (__bf16*)(ws + 23068672);     // 1024*1024*2  =  2 MiB
  __bf16* QKV = (__bf16*)(ws + 25165824);     // 8192*3072*2  = 48 MiB
  __bf16* Vt  = (__bf16*)(ws + 75497472);     // 64*64*2048*2 = 16 MiB
  __bf16* Ob  = (__bf16*)(ws + 92274688);     // 8192*1024*2  = 16 MiB
  // total: 109,051,904 bytes (~104 MiB)

  // 1) convert to bf16
  cvt_f32_bf16<<<(MTOK * DMODEL) / 256, 256, 0, stream>>>(x, Xb, MTOK * DMODEL);
  cvt_f32_bf16<<<(FQKV * DMODEL) / 256, 256, 0, stream>>>(wqkv, Wq, FQKV * DMODEL);
  cvt_f32_bf16<<<(DMODEL * DMODEL) / 256, 256, 0, stream>>>(wout, Wo, DMODEL * DMODEL);

  // 2) QKV projection: [8192,1024] x [3072,1024]^T -> bf16 [8192,3072]
  //    (0.125 attention scale folded into Q columns)
  gemm_bf16nt<0><<<dim3(FQKV / 64, MTOK / 128), 256, 0, stream>>>(
      Xb, Wq, (void*)QKV, DMODEL, DMODEL, DMODEL, FQKV);

  // 3) V transpose -> [B*H, D, T]
  transpose_v<<<dim3(SEQ / 256, HDIM, BATCH * NHEADS), 256, 0, stream>>>(QKV, Vt);

  // 4) causal flash attention -> bf16 [8192,1024]
  attn_fa<<<dim3(SEQ / 64, BATCH * NHEADS), 128, 0, stream>>>(QKV, Vt, Ob);

  // 5) output projection: [8192,1024] x [1024,1024]^T -> f32 d_out
  gemm_bf16nt<1><<<dim3(DMODEL / 64, MTOK / 128), 256, 0, stream>>>(
      Ob, Wo, (void*)out, DMODEL, DMODEL, DMODEL, DMODEL);
}
